// XConv_54022098649712
// MI455X (gfx1250) — compile-verified, matrix operand order checked by reading, not run on hardware
//
#include <hip/hip_runtime.h>

typedef __attribute__((ext_vector_type(16))) _Float16 v16h;
typedef __attribute__((ext_vector_type(8)))  _Float16 v8h;
typedef __attribute__((ext_vector_type(8)))  float    v8f;

#define B_    4
#define N_    8192
#define M_    2048
#define KNN_K 16
#define CIN   64
#define COUT  128

#define V8F_ZERO {0.f,0.f,0.f,0.f,0.f,0.f,0.f,0.f}

// ============================= FPS =============================
__global__ __launch_bounds__(256) void fps_kernel(const float* __restrict__ points,
                                                  float* __restrict__ sp_ws,
                                                  float* __restrict__ out_sp)
{
    const int b = blockIdx.x;
    const int t = threadIdx.x;
    __shared__ float dist[N_];
    __shared__ float rv[256];
    __shared__ int   ri[256];
    __shared__ int   s_far;
    __shared__ float s_cx, s_cy, s_cz;

    const float* P = points + (size_t)b * N_ * 3;

    for (int i = t; i < N_; i += 256) dist[i] = 1.0e10f;
    if (t == 0) { s_far = 0; s_cx = P[0]; s_cy = P[1]; s_cz = P[2]; }
    __syncthreads();

    for (int it = 0; it < M_; ++it) {
        const float cx = s_cx, cy = s_cy, cz = s_cz;
        if (t == 0) {
            size_t o = ((size_t)b * M_ + it) * 3;
            sp_ws[o+0] = cx; sp_ws[o+1] = cy; sp_ws[o+2] = cz;
            out_sp[o+0] = cx; out_sp[o+1] = cy; out_sp[o+2] = cz;
        }
        float bv = -1.0f; int bidx = 0;
        const int base = t * (N_ / 256);
        for (int j = 0; j < N_ / 256; ++j) {
            int i = base + j;
            float dx = P[i*3+0]-cx, dy = P[i*3+1]-cy, dz = P[i*3+2]-cz;
            float d  = dx*dx + dy*dy + dz*dz;
            float dm = fminf(dist[i], d);
            dist[i] = dm;
            if (dm > bv) { bv = dm; bidx = i; }
        }
        rv[t] = bv; ri[t] = bidx;
        __syncthreads();
        for (int s = 128; s > 0; s >>= 1) {
            if (t < s) { if (rv[t+s] > rv[t]) { rv[t] = rv[t+s]; ri[t] = ri[t+s]; } }
            __syncthreads();
        }
        if (t == 0) {
            int f = ri[0];
            s_far = f;
            s_cx = P[f*3+0]; s_cy = P[f*3+1]; s_cz = P[f*3+2];
        }
        __syncthreads();
    }
}

// ============================= kNN =============================
__global__ __launch_bounds__(256) void knn_kernel(const float* __restrict__ points,
                                                  const float* __restrict__ sp,
                                                  int* __restrict__ idx_ws,
                                                  float* __restrict__ out_idx)
{
    const int tid = blockIdx.x * blockDim.x + threadIdx.x;
    if (tid >= B_ * M_) return;
    const int b = tid / M_;
    const float* P = points + (size_t)b * N_ * 3;
    const float qx = sp[(size_t)tid*3+0], qy = sp[(size_t)tid*3+1], qz = sp[(size_t)tid*3+2];

    float bd[KNN_K]; int bi[KNN_K];
#pragma unroll
    for (int j = 0; j < KNN_K; ++j) { bd[j] = 3.0e38f; bi[j] = 0; }

    for (int n = 0; n < N_; ++n) {
        float dx = P[n*3+0]-qx, dy = P[n*3+1]-qy, dz = P[n*3+2]-qz;
        float d  = dx*dx + dy*dy + dz*dz;
        if (d < bd[KNN_K-1]) {
            bd[KNN_K-1] = d; bi[KNN_K-1] = n;
#pragma unroll
            for (int j = KNN_K-1; j > 0; --j) {
                if (bd[j] < bd[j-1]) {
                    float td = bd[j]; bd[j] = bd[j-1]; bd[j-1] = td;
                    int   ti = bi[j]; bi[j] = bi[j-1]; bi[j-1] = ti;
                }
            }
        }
    }
#pragma unroll
    for (int j = 0; j < KNN_K; ++j) {
        idx_ws[(size_t)tid*KNN_K + j]  = bi[j];
        out_idx[(size_t)tid*KNN_K + j] = (float)bi[j];
    }
}

// ===================== weight repack (f32 -> f16 B-layout) =====================
// Packed: v16h tile per (ntile, kchunk, lane); lane l holds col l%16, halves e ->
// K = kc*32 + (e>>3)*16 + (l>>4)*8 + (e&7).  Zero-pads K >= Kreal.
__global__ __launch_bounds__(256) void repack_kernel(const float* __restrict__ W,
                                                     _Float16* __restrict__ outp,
                                                     int Kreal, int nkc, int Ncols)
{
    const int idx = blockIdx.x * 256 + threadIdx.x;           // (tile,kc,lane)
    const int total = (Ncols / 16) * nkc * 32;
    if (idx >= total) return;
    const int lane = idx & 31;
    const int kc   = (idx >> 5) % nkc;
    const int tile = (idx >> 5) / nkc;
    const int col  = lane & 15, g = lane >> 4;
    v16h v;
#pragma unroll
    for (int e = 0; e < 16; ++e) {
        int k = kc*32 + ((e >> 3) << 4) + (g << 3) + (e & 7);
        float f = (k < Kreal) ? W[k * Ncols + tile*16 + col] : 0.0f;
        v[e] = (_Float16)f;
    }
    ((v16h*)outp)[idx] = v;
}

// ===================== WMMA helpers =====================
// A tile (16x32 f16) from f16 LDS row: two contiguous 16B segments.
__device__ __forceinline__ v16h load_a16(const _Float16* p, int g)
{
    const v8h lo = *(const v8h*)(p + (g << 3));
    const v8h hi = *(const v8h*)(p + 16 + (g << 3));
    v16h a;
#pragma unroll
    for (int e = 0; e < 8; ++e) { a[e] = lo[e]; a[e + 8] = hi[e]; }
    return a;
}

// D = act(A(16xKD f16 LDS) @ PW(packed) + bias) -> f16 LDS row-major [16 x NC]
template<int KD, int NC, bool RELU>
__device__ __forceinline__ void gemm_f16(const _Float16* A,
                                         const v16h* __restrict__ PW,
                                         const float* __restrict__ bias,
                                         _Float16* D, int lane)
{
    const int row = lane & 15, g = lane >> 4;
    constexpr int NKC = KD / 32;
#pragma unroll
    for (int t = 0; t < NC / 16; ++t) {
        v8f acc = V8F_ZERO;
#pragma unroll
        for (int kc = 0; kc < NKC; ++kc) {
            v16h a  = load_a16(A + row * KD + kc * 32, g);
            v16h bt = PW[(t * NKC + kc) * 32 + lane];
            acc = __builtin_amdgcn_wmma_f32_16x16x32_f16(false, a, false, bt,
                                                         (short)0, acc, false, false);
        }
        float bb = bias[t*16 + row];
#pragma unroll
        for (int i = 0; i < 8; ++i) {
            float v = acc[i] + bb;
            if (RELU) v = fmaxf(v, 0.0f);
            D[(i + (g << 3)) * NC + t*16 + row] = (_Float16)v;   // D: row=i+8g, col=lane%16
        }
    }
}

// ========================= XConv main =========================
__global__ __launch_bounds__(32) void xconv_kernel(
    const float* __restrict__ points,  const float* __restrict__ features,
    const float* __restrict__ sp,      const int*   __restrict__ knn,
    const v16h* __restrict__ pw1,  const float* __restrict__ b1,
    const v16h* __restrict__ pw2,  const float* __restrict__ b2,
    const v16h* __restrict__ pw3,  const float* __restrict__ b3,
    const v16h* __restrict__ pcw1, const float* __restrict__ cb1,
    const v16h* __restrict__ pcw2, const float* __restrict__ cb2,
    const float* __restrict__ ln_g, const float* __restrict__ ln_b,
    float* __restrict__ out_o)
{
    const int bm   = blockIdx.x;       // b*M_ + m
    const int b    = bm / M_;
    const int lane = threadIdx.x;
    const int row  = lane & 15, g = lane >> 4;

    __shared__ _Float16 aP[16 * 32];     // neigh pts, K-padded 3->32
    __shared__ _Float16 aH1[16 * 64];    // H1, reused for TF
    __shared__ _Float16 aH2[16 * 64];
    __shared__ _Float16 aXMT[16 * 32];   // XM^T, K-padded 16->32
    __shared__ _Float16 aO1[16 * 128];
    __shared__ v16h     bF[4 * 32];      // gathered features in B-register layout
    __shared__ float    sOMAX[COUT];

    const float* P  = points   + (size_t)b * N_ * 3;
    const float* F  = features + (size_t)b * N_ * CIN;
    const int*   nb = knn + (size_t)bm * KNN_K;
    const float qx = sp[(size_t)bm*3+0], qy = sp[(size_t)bm*3+1], qz = sp[(size_t)bm*3+2];

    // zero-fill padded A buffers
    for (int q = lane; q < 16 * 32; q += 32) {
        aP[q]   = (_Float16)0.0f;
        aXMT[q] = (_Float16)0.0f;
    }
    __syncthreads();

    // gather centered neighbor points (rows 0..15, cols 0..2)
    if (lane < 16) {
        int n = nb[lane];
        aP[lane*32 + 0] = (_Float16)(P[n*3+0] - qx);
        aP[lane*32 + 1] = (_Float16)(P[n*3+1] - qy);
        aP[lane*32 + 2] = (_Float16)(P[n*3+2] - qz);
    }

    // gather neighbor features directly into B-register layout (K padded 16->32)
    {
        int nidx[8];
#pragma unroll
        for (int e = 0; e < 8; ++e) nidx[e] = nb[(g << 3) + e];
#pragma unroll
        for (int t = 0; t < 4; ++t) {
            v16h v;
#pragma unroll
            for (int e = 0; e < 8; ++e) {
                v[e]     = (_Float16)F[(size_t)nidx[e] * CIN + t*16 + row];
                v[e + 8] = (_Float16)0.0f;
            }
            bF[t*32 + lane] = v;
        }
    }
    __syncthreads();

    // stage1: H1 = relu(P @ w1 + b1)        16x32(3)x64
    gemm_f16<32, 64, true>(aP, pw1, b1, aH1, lane);
    __syncthreads();
    // stage2: H2 = relu(H1 @ w2 + b2)       16x64x64
    gemm_f16<64, 64, true>(aH1, pw2, b2, aH2, lane);
    __syncthreads();
    // stage3: XM = mean_rows(H2 @ w3 + b3)  16x64x256 -> write XM^T (f16, K-padded)
#pragma unroll
    for (int t = 0; t < 16; ++t) {
        v8f acc = V8F_ZERO;
#pragma unroll
        for (int kc = 0; kc < 2; ++kc) {
            v16h a  = load_a16(aH2 + row*64 + kc*32, g);
            v16h bt = pw3[(t*2 + kc)*32 + lane];
            acc = __builtin_amdgcn_wmma_f32_16x16x32_f16(false, a, false, bt,
                                                         (short)0, acc, false, false);
        }
        float s8 = 0.0f;
#pragma unroll
        for (int i = 0; i < 8; ++i) s8 += acc[i];
        float s16 = s8 + __shfl_xor(s8, 16, 32);
        int   n   = t*16 + row;                       // index into xm mean vec [256]
        float meanv = s16 * (1.0f / 16.0f) + b3[n];
        if (g == 0) aXMT[(n & 15) * 32 + (n >> 4)] = (_Float16)meanv;  // XM^T[j][k]
    }
    __syncthreads();
    // stage4: TF = XM^T @ F                 16x32(16)x64 -> aH1
#pragma unroll
    for (int t = 0; t < 4; ++t) {
        v8f acc = V8F_ZERO;
        v16h a  = load_a16(aXMT + row*32, g);
        v16h bt = bF[t*32 + lane];
        acc = __builtin_amdgcn_wmma_f32_16x16x32_f16(false, a, false, bt,
                                                     (short)0, acc, false, false);
#pragma unroll
        for (int i = 0; i < 8; ++i)
            aH1[(i + (g << 3)) * 64 + t*16 + row] = (_Float16)acc[i];
    }
    __syncthreads();
    // stage5: O1 = relu(TF @ cw1 + cb1)     16x64x128
    gemm_f16<64, 128, true>(aH1, pcw1, cb1, aO1, lane);
    __syncthreads();
    // stage6: O2 = O1 @ cw2 + cb2; column max over 16 rows   16x128x128
#pragma unroll
    for (int t = 0; t < 8; ++t) {
        v8f acc = V8F_ZERO;
#pragma unroll
        for (int kc = 0; kc < 4; ++kc) {
            v16h a  = load_a16(aO1 + row*128 + kc*32, g);
            v16h bt = pcw2[(t*4 + kc)*32 + lane];
            acc = __builtin_amdgcn_wmma_f32_16x16x32_f16(false, a, false, bt,
                                                         (short)0, acc, false, false);
        }
        float mx = acc[0];
#pragma unroll
        for (int i = 1; i < 8; ++i) mx = fmaxf(mx, acc[i]);
        mx = fmaxf(mx, __shfl_xor(mx, 16, 32));
        if (g == 0) sOMAX[t*16 + row] = mx + cb2[t*16 + row];
    }
    __syncthreads();

    // layernorm over 128 channels; each lane owns 4 channels
    const int c0 = lane * 4;
    float v0 = sOMAX[c0+0], v1 = sOMAX[c0+1], v2 = sOMAX[c0+2], v3 = sOMAX[c0+3];
    float s = v0 + v1 + v2 + v3;
#pragma unroll
    for (int off = 16; off > 0; off >>= 1) s += __shfl_xor(s, off, 32);
    float mu = s * (1.0f / 128.0f);
    float d0 = v0-mu, d1 = v1-mu, d2 = v2-mu, d3 = v3-mu;
    float sv = d0*d0 + d1*d1 + d2*d2 + d3*d3;
#pragma unroll
    for (int off = 16; off > 0; off >>= 1) sv += __shfl_xor(sv, off, 32);
    float inv = rsqrtf(sv * (1.0f / 128.0f) + 1e-5f);
    float* O = out_o + (size_t)bm * COUT;
    O[c0+0] = d0 * inv * ln_g[c0+0] + ln_b[c0+0];
    O[c0+1] = d1 * inv * ln_g[c0+1] + ln_b[c0+1];
    O[c0+2] = d2 * inv * ln_g[c0+2] + ln_b[c0+2];
    O[c0+3] = d3 * inv * ln_g[c0+3] + ln_b[c0+3];
}

// ========================= launcher =========================
extern "C" void kernel_launch(void* const* d_in, const int* in_sizes, int n_in,
                              void* d_out, int out_size, void* d_ws, size_t ws_size,
                              hipStream_t stream)
{
    const float* points   = (const float*)d_in[0];
    const float* features = (const float*)d_in[1];
    const float* w1  = (const float*)d_in[2];
    const float* b1  = (const float*)d_in[3];
    const float* w2  = (const float*)d_in[4];
    const float* b2  = (const float*)d_in[5];
    const float* w3  = (const float*)d_in[6];
    const float* b3  = (const float*)d_in[7];
    const float* cw1 = (const float*)d_in[8];
    const float* cb1 = (const float*)d_in[9];
    const float* cw2 = (const float*)d_in[10];
    const float* cb2 = (const float*)d_in[11];
    const float* ln_g = (const float*)d_in[12];
    const float* ln_b = (const float*)d_in[13];

    float* out     = (float*)d_out;
    float* out_sp  = out;                                        // B*M*3
    float* out_o   = out + (size_t)B_*M_*3;                      // B*M*COUT
    float* out_idx = out + (size_t)B_*M_*3 + (size_t)B_*M_*COUT; // B*M*K

    // workspace layout (bytes)
    char* ws = (char*)d_ws;
    float*     sp_ws  = (float*)ws;                              // 24576 f32
    int*       idx_ws = (int*)(ws + 98304);                      // 131072 i32
    _Float16*  pw1    = (_Float16*)(ws + 622592);                // 2048  f16 (32x64)
    _Float16*  pw2    = (_Float16*)(ws + 622592 + 4096);         // 4096  f16 (64x64)
    _Float16*  pw3    = (_Float16*)(ws + 622592 + 12288);        // 16384 f16 (64x256)
    _Float16*  pcw1   = (_Float16*)(ws + 622592 + 45056);        // 8192  f16 (64x128)
    _Float16*  pcw2   = (_Float16*)(ws + 622592 + 61440);        // 16384 f16 (128x128)

    // one-time (per call) weight repack into WMMA B-register layout
    repack_kernel<<<1, 256, 0, stream>>>(w1,  pw1,  3,   1,  64);   // 4*1*32 = 128
    repack_kernel<<<1, 256, 0, stream>>>(w2,  pw2,  64,  2,  64);   // 4*2*32 = 256
    repack_kernel<<<4, 256, 0, stream>>>(w3,  pw3,  64,  2, 256);   // 16*2*32 = 1024
    repack_kernel<<<2, 256, 0, stream>>>(cw1, pcw1, 64,  2, 128);   // 8*2*32 = 512
    repack_kernel<<<4, 256, 0, stream>>>(cw2, pcw2, 128, 4, 128);   // 8*4*32 = 1024

    fps_kernel<<<B_, 256, 0, stream>>>(points, sp_ws, out_sp);
    knn_kernel<<<(B_*M_ + 255)/256, 256, 0, stream>>>(points, sp_ws, idx_ws, out_idx);
    xconv_kernel<<<B_*M_, 32, 0, stream>>>(points, features, sp_ws, idx_ws,
                                           (const v16h*)pw1, b1, (const v16h*)pw2, b2,
                                           (const v16h*)pw3, b3, (const v16h*)pcw1, cb1,
                                           (const v16h*)pcw2, cb2, ln_g, ln_b, out_o);
}
